// Dcls2d_4398046511417
// MI455X (gfx1250) — matrix operand order repaired and checked by reading, not compiled, not run on hardware
//
#include <hip/hip_runtime.h>
#include <stdint.h>

typedef __bf16 bf16;
typedef bf16  v8bf  __attribute__((ext_vector_type(8)));
typedef bf16  v16bf __attribute__((ext_vector_type(16)));
typedef float v8f   __attribute__((ext_vector_type(8)));

#define HK   7
#define WK   7
#define CIN  128
#define COUT 128
#define NIMG 32
#define HH   56
#define WW   56
#define HP   62      // 56 + 2*3 halo
#define WP   64      // 56 + 2*3 halo, rounded to 64
#define KPTS 9
#define NPIX (HH*WW) // 3136
#define TAPS (HK*WK) // 49
#define LDA  40      // LDS row stride (elems) for 32-wide c tiles: 80B -> 16B aligned, bank-staggered
#define LDT  72      // transpose-tile LDS stride

#define USE_ASYNC 1  // gfx1250 GLOBAL_LOAD_ASYNC_TO_LDS_B128 path (set 0 to fall back)

// ------------- Kernel 1: NCHW fp32 -> padded NHWC bf16 (LDS-tiled transpose) -------------
// xpad[n][yp][xp][c] = x[n][c][yp-3][xp-3] (0 in halo).  One (n,yp) plane per block.
__global__ __launch_bounds__(256)
void nhwc_pad_kernel(const float* __restrict__ xin, bf16* __restrict__ xpad) {
    __shared__ bf16 T[CIN * LDT];           // [c][xp]
    const int yp = blockIdx.x;              // 0..61
    const int n  = blockIdx.y;
    const int t  = threadIdx.x;
    const int y  = yp - 3;
    const bool yok = (unsigned)y < HH;
    {
        int c  = t >> 1;
        int xh = (t & 1) * 32;
        const float* row = xin + (((size_t)n * CIN + c) * HH + (yok ? y : 0)) * WW;
#pragma unroll
        for (int i = 0; i < 32; ++i) {
            int xp = xh + i;
            int x  = xp - 3;
            float v = (yok && (unsigned)x < WW) ? row[x] : 0.0f;
            T[c * LDT + xp] = (bf16)v;
        }
    }
    __syncthreads();
    {
        int xp = t >> 2;                    // 0..63
        int cq = (t & 3) * 32;
        union { bf16 e[32]; uint4 q[4]; } u;
#pragma unroll
        for (int j = 0; j < 32; ++j) u.e[j] = T[(cq + j) * LDT + xp];
        uint4* dst = (uint4*)(xpad + ((((size_t)n * HP + yp) * WP + xp) * CIN + cq));
        dst[0] = u.q[0]; dst[1] = u.q[1]; dst[2] = u.q[2]; dst[3] = u.q[3];
    }
}

// ------------- Kernel 2: bilinear scatter -> dense bf16 kernel, layout [tap][o][c] -------
__global__ void build_kernel(const float* __restrict__ weight, const float* __restrict__ P,
                             bf16* __restrict__ Bmat) {
    int t = blockIdx.x * blockDim.x + threadIdx.x;    // < COUT*CIN
    int c = t & (CIN - 1);
    int o = t >> 7;
    float kern[TAPS];
#pragma unroll
    for (int i = 0; i < TAPS; ++i) kern[i] = 0.0f;
    for (int k = 0; k < KPTS; ++k) {
        float ph = fminf(fmaxf(P[(0 * CIN + c) * KPTS + k], -3.0f), 3.0f) + 3.0f;  // [0,6]
        float pw = fminf(fmaxf(P[(1 * CIN + c) * KPTS + k], -3.0f), 3.0f) + 3.0f;  // [0,6]
        int ih = (int)floorf(ph);
        int iw = (int)floorf(pw);
        float rh = ph - (float)ih;
        float rw = pw - (float)iw;
        float w = weight[((size_t)o * CIN + c) * KPTS + k];
        float w00 = w * (1.0f - rh) * (1.0f - rw);
        float w01 = w * (1.0f - rh) * rw;
        float w10 = w * rh * (1.0f - rw);
        float w11 = w * rh * rw;
        kern[ih * WK + iw] += w00;                                   // ih,iw <= 6 always
        if (iw + 1 < WK)                kern[ih * WK + iw + 1]       += w01;
        if (ih + 1 < HK)                kern[(ih + 1) * WK + iw]     += w10;
        if (ih + 1 < HK && iw + 1 < WK) kern[(ih + 1) * WK + iw + 1] += w11;
    }
#pragma unroll
    for (int tap = 0; tap < TAPS; ++tap)
        Bmat[(size_t)tap * COUT * CIN + (size_t)o * CIN + c] = (bf16)kern[tap];
}

// ------------- Kernel 3: implicit-GEMM conv via bf16 WMMA ---------------------------------
// Block = 256 threads = 8 waves. Tile: 128 O x 64 pixels. Wave w: O rows [16w,16w+16).
// A frags: direct global b128 loads (rows disjoint per wave -> no sharing, no LDS).
// B tile : 64 pix x 32 c staged NHWC->LDS via async b128 loads (ASYNCcnt), barrier, 4 WMMAs.
__global__ __launch_bounds__(256)
void conv_wmma(const bf16* __restrict__ xpad, const bf16* __restrict__ Bmat,
               const float* __restrict__ bias, float* __restrict__ out) {
    __shared__ bf16 Xs[64 * LDA];     // image tile: 64 pix x 32 c (+pad)

    const int tid   = threadIdx.x;
    const int wv    = tid >> 5;
    const int lane  = tid & 31;
    const int hf    = lane >> 4;      // half-wave selects K sub-range per ISA layout
    const int lr    = lane & 15;
    const int oW    = wv * 16;
    const int n     = blockIdx.y;
    const int pBase = blockIdx.x * 64;

    // Accumulators pre-loaded with bias (D VGPR j <-> row o = oW + hf*8 + j)
    v8f acc[4];
#pragma unroll
    for (int j = 0; j < 8; ++j) {
        float bv = bias[oW + hf * 8 + j];
        acc[0][j] = bv; acc[1][j] = bv; acc[2][j] = bv; acc[3][j] = bv;
    }

    // Per-thread X staging geometry (one b128 = 8 c-values of one pixel per iteration)
    const int prow = tid >> 2;                  // pixel within tile 0..63
    const int cq   = (tid & 3) * 8;             // c sub-offset 0/8/16/24
    {
        // nothing
    }
    int p = pBase + prow;
    int y = p / WW;
    int x = p - y * WW;
    const bf16* xbase = xpad + ((((size_t)n * HP + y) * WP + x) * CIN + cq);
    const uint32_t ldsOff = (uint32_t)(uintptr_t)&Xs[prow * LDA + cq];

    // Per-lane A fragment base (row oW+lr of Bmat, K-contiguous)
    const bf16* aRow = Bmat + (size_t)(oW + lr) * CIN;

    for (int tap = 0; tap < TAPS; ++tap) {
        const int kh = tap / WK, kw = tap % WK;
        if (tap + 1 < TAPS)   // pull next tap's kernel tile toward the caches
            __builtin_prefetch(Bmat + (size_t)(tap + 1) * COUT * CIN + tid * 64, 0, 0);
        const size_t xTapOff = (size_t)(kh * WP + kw) * CIN;
        const bf16*  aTap    = aRow + (size_t)tap * COUT * CIN;

        for (int cb = 0; cb < 4; ++cb) {
            const int c0 = cb * 32;
            __syncthreads();   // previous iteration done reading Xs
            // ---- issue async B-tile stage: 64 pix x 32 c, one b128 per thread ----
            const bf16* gsrc = xbase + xTapOff + c0;
#if USE_ASYNC
            asm volatile("global_load_async_to_lds_b128 %0, %1, off"
                         : : "v"(ldsOff), "v"((uint64_t)(uintptr_t)gsrc) : "memory");
#else
            {
                uint4 d = *(const uint4*)gsrc;
                *(uint4*)&Xs[prow * LDA + cq] = d;
            }
#endif
            // ---- A fragment direct from global (overlaps the async copy) ----
            // half 0 holds K 0-7|16-23, half 1 holds K 8-15|24-31 (ISA 16-bit A layout)
            v8bf alo = *(const v8bf*)(aTap + c0 + (hf ? 8  : 0));
            v8bf ahi = *(const v8bf*)(aTap + c0 + (hf ? 24 : 16));
            v16bf a;
#pragma unroll
            for (int i = 0; i < 8; ++i) { a[i] = alo[i]; a[8 + i] = ahi[i]; }

#if USE_ASYNC
            asm volatile("s_wait_asynccnt 0" : : : "memory");
#endif
            __syncthreads();   // B tile visible to all waves

#pragma unroll
            for (int pT = 0; pT < 4; ++pT) {
                // B fragment: column pixel = pT*16+lr; half 0 = K 0-7|16-23, half 1 = K 8-15|24-31
                v8bf blo = *(const v8bf*)&Xs[(pT * 16 + lr) * LDA + hf * 8];
                v8bf bhi = *(const v8bf*)&Xs[(pT * 16 + lr) * LDA + 16 + hf * 8];
                v16bf b;
#pragma unroll
                for (int i = 0; i < 8; ++i) { b[i] = blo[i]; b[8 + i] = bhi[i]; }
                acc[pT] = __builtin_amdgcn_wmma_f32_16x16x32_bf16(
                    false, a, false, b, (short)0, acc[pT], false, false);
            }
        }
    }

    // ---- writeout: lanes 0-15 / 16-31 write contiguous 64B runs of one channel ----
    float* outn = out + (size_t)n * COUT * NPIX;
#pragma unroll
    for (int pT = 0; pT < 4; ++pT) {
#pragma unroll
        for (int j = 0; j < 8; ++j) {
            int o = oW + hf * 8 + j;
            outn[(size_t)o * NPIX + pBase + pT * 16 + lr] = acc[pT][j];
        }
    }
}

// ---------------------------- host launcher ----------------------------------
extern "C" void kernel_launch(void* const* d_in, const int* in_sizes, int n_in,
                              void* d_out, int out_size, void* d_ws, size_t ws_size,
                              hipStream_t stream) {
    (void)in_sizes; (void)n_in; (void)out_size; (void)ws_size;
    const float* x      = (const float*)d_in[0];
    const float* weight = (const float*)d_in[1];
    const float* P      = (const float*)d_in[2];
    const float* bias   = (const float*)d_in[3];
    float* out = (float*)d_out;

    char* ws = (char*)d_ws;
    const size_t xpadElems = (size_t)NIMG * HP * WP * CIN;            // 16,252,928 (NHWC)
    bf16* xpad = (bf16*)ws;
    size_t bmOff = (xpadElems * sizeof(bf16) + 255) & ~(size_t)255;
    bf16* Bmat = (bf16*)(ws + bmOff);                                  // 49*128*128 bf16

    dim3 gt(HP, NIMG);                                                 // (62, 32)
    nhwc_pad_kernel<<<gt, 256, 0, stream>>>(x, xpad);
    build_kernel<<<(COUT * CIN) / 256, 256, 0, stream>>>(weight, P, Bmat);
    dim3 g(NPIX / 64, NIMG);                                           // (49, 32)
    conv_wmma<<<g, 256, 0, stream>>>(xpad, Bmat, bias, out);
}
